// IGMC_23751169146882
// MI455X (gfx1250) — matrix operand order repaired and checked by reading, not compiled, not run on hardware
//
#include <hip/hip_runtime.h>
#include <hip/hip_fp16.h>
#include <math.h>

typedef __attribute__((ext_vector_type(16))) _Float16 v16h;
typedef __attribute__((ext_vector_type(8)))  _Float16 v8h;
typedef __attribute__((ext_vector_type(8)))  float    v8f;

// ---------------------------------------------------------------------------
// Layer-0 input: pad x [N,4] f32 into hf16 [N,32] f16 (zeros beyond col 3).
// Also zero the f16 atomic aggregation buffer (fresh every call; no reliance
// on workspace state, no memset node needed in the captured graph).
// ---------------------------------------------------------------------------
__global__ void igmc_init_h0(const float* __restrict__ x,
                             _Float16* __restrict__ hf16,
                             _Float16* __restrict__ agg16, int nNodes) {
  int t = blockIdx.x * blockDim.x + threadIdx.x;
  if (t >= nNodes * 32) return;
  int n = t >> 5, o = t & 31;
  hf16[t]  = (_Float16)(o < 4 ? x[n * 4 + o] : 0.0f);
  agg16[t] = (_Float16)0.0f;
}

// ---------------------------------------------------------------------------
// Build per-layer weights, transposed: W[mat][o][k] f16, mat 0..4 = relations
// (sum_b comp[r,b]*bases[b,k,o]), mat 5 = self-loop weight. k >= Kin -> 0.
// ---------------------------------------------------------------------------
__global__ void igmc_prep_weights(const float* __restrict__ bases,
                                  const float* __restrict__ comp,
                                  const float* __restrict__ loopw,
                                  int Kin, _Float16* __restrict__ W) {
  int t = blockIdx.x * blockDim.x + threadIdx.x;   // 6*32*32 = 6144
  if (t >= 6144) return;
  int mat = t >> 10;
  int o   = (t >> 5) & 31;
  int k   = t & 31;
  float v = 0.0f;
  if (k < Kin) {
    if (mat < 5)
      v = comp[mat * 2 + 0] * bases[(0 * Kin + k) * 32 + o] +
          comp[mat * 2 + 1] * bases[(1 * Kin + k) * 32 + o];
    else
      v = loopw[k * 32 + o];
  }
  W[t] = (_Float16)v;   // stored [mat][o][k]
}

// ---------------------------------------------------------------------------
// Fused WMMA GEMM: one wave owns a 16-row node tile, loads its A fragment
// once, then chains 12 WMMAs (6 matrices x 2 col tiles, K=32 each).
// mats 0..4 -> hr (f16, coalesced b128 stores via per-wave LDS transpose),
// mat 5     -> aggLoop = h @ loop_w + bias (f32).
// ---------------------------------------------------------------------------
__global__ void igmc_gemm_hr(const _Float16* __restrict__ hf16, // [N,32]
                             const _Float16* __restrict__ W,    // [6,32(o),32(k)]
                             const float* __restrict__ bias,    // [32]
                             _Float16* __restrict__ hr,         // [5,N,32]
                             float* __restrict__ aggLoop,       // [N,32]
                             int nNodes) {
  __shared__ _Float16 lds[8 * 256];                // 512B tile per wave
  int wave = (blockIdx.x * blockDim.x + threadIdx.x) >> 5;
  int lane = threadIdx.x & 31;
  int w    = (threadIdx.x >> 5) & 7;
  int tilesM = nNodes >> 4;
  if (wave >= tilesM) return;                      // wave-uniform exit
  int g  = lane >> 4;
  int lr = lane & 15;

  // A fragment: row (wave*16+lr); half j holds k = (j&7) + 16*(j>>3) + 8*g
  const _Float16* arow = hf16 + (size_t)((wave << 4) + lr) * 32;
  v8h a0 = *(const v8h*)(arow + (g << 3));
  v8h a1 = *(const v8h*)(arow + 16 + (g << 3));
  v16h a = __builtin_shufflevector(a0, a1, 0,1,2,3,4,5,6,7,8,9,10,11,12,13,14,15);

  int r  = lane >> 1;                              // LDS read row
  int hh = lane & 1;                               // LDS read half-row

#pragma unroll
  for (int mat = 0; mat < 6; ++mat) {
#pragma unroll
    for (int nt = 0; nt < 2; ++nt) {
      int n = (nt << 4) + lr;
      // B fragment: half j holds k = (j&7) + 16*(j>>3) + 8*g, in W[mat][n][k]
      const _Float16* bp = W + mat * 1024 + n * 32 + (g << 3);
      v8h b0 = *(const v8h*)(bp);
      v8h b1 = *(const v8h*)(bp + 16);
      v16h b = __builtin_shufflevector(b0, b1, 0,1,2,3,4,5,6,7,8,9,10,11,12,13,14,15);

      v8f c = {};
      c = __builtin_amdgcn_wmma_f32_16x16x32_f16(false, a, false, b,
                                                 (short)0, c, false, false);

      if (mat < 5) {
        // Transpose tile in LDS (same-wave DS ops are in-order; no barrier)
#pragma unroll
        for (int i = 0; i < 8; ++i)
          lds[w * 256 + (i + (g << 3)) * 16 + lr] = (_Float16)c[i];
        v8h vr = *(const v8h*)&lds[w * 256 + r * 16 + hh * 8];
        _Float16* outp = hr + (size_t)mat * nNodes * 32 +
                         (size_t)((wave << 4) + r) * 32 + (nt << 4) + hh * 8;
        *(v8h*)outp = vr;                          // coalesced 16B store
      } else {
        float bv = bias[n];
#pragma unroll
        for (int i = 0; i < 8; ++i)
          aggLoop[(size_t)((wave << 4) + i + (g << 3)) * 32 + n] = c[i] + bv;
      }
    }
  }
}

// ---------------------------------------------------------------------------
// Edge scatter: agg16[dst] += hr[etype, src] with packed-f16 atomics.
// 4 lanes/edge: one 16B load + 4 global_atomic_pk_add_f16 each
// (half the atomic ops and traffic of the f32 path; fine under tanh).
// ---------------------------------------------------------------------------
__global__ void igmc_edge_scatter(const int* __restrict__ src,
                                  const int* __restrict__ dst,
                                  const int* __restrict__ et,
                                  const _Float16* __restrict__ hr,
                                  _Float16* __restrict__ agg16,
                                  int nEdges, int nNodes) {
  long long t = (long long)blockIdx.x * blockDim.x + threadIdx.x;
  int e = (int)(t >> 2);
  int c = (int)(t & 3);
  if (e >= nEdges) return;
  int s = src[e], d = dst[e], r = et[e];
  uint4 mv = *(const uint4*)(hr + ((size_t)r * nNodes + s) * 32 + c * 8);
  __half2* ap = (__half2*)(agg16 + (size_t)d * 32 + c * 8);
  atomicAdd(ap + 0, __builtin_bit_cast(__half2, mv.x));
  atomicAdd(ap + 1, __builtin_bit_cast(__half2, mv.y));
  atomicAdd(ap + 2, __builtin_bit_cast(__half2, mv.z));
  atomicAdd(ap + 3, __builtin_bit_cast(__half2, mv.w));
}

// ---------------------------------------------------------------------------
// h = tanh(selfloop_f32 + messages_f16); write f32 concat slice + f16 input.
// Re-zeroes agg16 after consuming it so the next layer's scatter starts clean
// (rides along with a pass already touching those lines; replaces memset).
// ---------------------------------------------------------------------------
__global__ void igmc_activate(const float* __restrict__ aggLoop,
                              _Float16* __restrict__ agg16,
                              float* __restrict__ concat,
                              _Float16* __restrict__ hf16,
                              int layer, int nNodes) {
  int t = blockIdx.x * blockDim.x + threadIdx.x;
  if (t >= nNodes * 32) return;
  int n = t >> 5, o = t & 31;
  float v = tanhf(aggLoop[t] + (float)agg16[t]);
  agg16[t] = (_Float16)0.0f;
  concat[(size_t)n * 128 + layer * 32 + o] = v;
  hf16[t] = (_Float16)v;
}

// ---------------------------------------------------------------------------
// W1 [256,128] f32 -> transposed f16 [128(o),256(k)]
// ---------------------------------------------------------------------------
__global__ void igmc_cvt_w1(const float* __restrict__ W1,
                            _Float16* __restrict__ W1t) {
  int t = blockIdx.x * blockDim.x + threadIdx.x;   // 32768
  if (t >= 256 * 128) return;
  int k = t >> 7, o = t & 127;
  W1t[o * 256 + k] = (_Float16)W1[t];
}

// ---------------------------------------------------------------------------
// Gather z = [concat[user] | concat[item]] -> f16 [4096,256]
// ---------------------------------------------------------------------------
__global__ void igmc_gather_z(const float* __restrict__ concat,
                              const int* __restrict__ uidx,
                              const int* __restrict__ iidx,
                              _Float16* __restrict__ z, int nB) {
  int t = blockIdx.x * blockDim.x + threadIdx.x;
  if (t >= nB * 256) return;
  int b = t >> 8, col = t & 255;
  int node = (col < 128) ? uidx[b] : iidx[b];
  z[t] = (_Float16)concat[(size_t)node * 128 + (col & 127)];
}

// ---------------------------------------------------------------------------
// MLP layer 1: z1 = relu(z @ W1 + b1), [4096,256]x[256,128], 8 chained WMMAs.
// ---------------------------------------------------------------------------
__global__ void igmc_mlp1(const _Float16* __restrict__ z,
                          const _Float16* __restrict__ W1t, // [128(o),256(k)]
                          const float* __restrict__ b1,
                          _Float16* __restrict__ z1, int nB) {
  int wave = (blockIdx.x * blockDim.x + threadIdx.x) >> 5;
  int lane = threadIdx.x & 31;
  int mt = wave >> 3;          // 256 row tiles
  int nt = wave & 7;           // 8 col tiles
  if (mt >= (nB >> 4)) return;
  int g  = lane >> 4;
  int lr = lane & 15;
  int n  = (nt << 4) + lr;
  const _Float16* arow = z + (size_t)((mt << 4) + lr) * 256;
  const _Float16* brow = W1t + (size_t)n * 256;
  v8f c = {};
#pragma unroll
  for (int kb = 0; kb < 8; ++kb) {
    int k0 = kb << 5;
    v8h a0 = *(const v8h*)(arow + k0 + (g << 3));
    v8h a1 = *(const v8h*)(arow + k0 + 16 + (g << 3));
    v16h a = __builtin_shufflevector(a0, a1, 0,1,2,3,4,5,6,7,8,9,10,11,12,13,14,15);
    v8h b0 = *(const v8h*)(brow + k0 + (g << 3));
    v8h b1 = *(const v8h*)(brow + k0 + 16 + (g << 3));
    v16h b = __builtin_shufflevector(b0, b1, 0,1,2,3,4,5,6,7,8,9,10,11,12,13,14,15);
    c = __builtin_amdgcn_wmma_f32_16x16x32_f16(false, a, false, b,
                                               (short)0, c, false, false);
  }
  float bv = b1[n];
#pragma unroll
  for (int i = 0; i < 8; ++i) {
    int m = (mt << 4) + i + (g << 3);
    float v = c[i] + bv;
    z1[(size_t)m * 128 + n] = (_Float16)(v > 0.0f ? v : 0.0f);
  }
}

// ---------------------------------------------------------------------------
// Head: logits = z1 @ W2 + b2 (128->5), then log_softmax. One thread per row.
// ---------------------------------------------------------------------------
__global__ void igmc_head(const _Float16* __restrict__ z1,
                          const float* __restrict__ W2,
                          const float* __restrict__ b2,
                          float* __restrict__ out, int nB) {
  int bi = blockIdx.x * blockDim.x + threadIdx.x;
  if (bi >= nB) return;
  float l[5];
#pragma unroll
  for (int c = 0; c < 5; ++c) l[c] = b2[c];
  const _Float16* zr = z1 + (size_t)bi * 128;
  for (int k = 0; k < 128; ++k) {
    float zv = (float)zr[k];
#pragma unroll
    for (int c = 0; c < 5; ++c) l[c] += zv * W2[k * 5 + c];
  }
  float mx = l[0];
#pragma unroll
  for (int c = 1; c < 5; ++c) mx = fmaxf(mx, l[c]);
  float s = 0.0f;
#pragma unroll
  for (int c = 0; c < 5; ++c) s += expf(l[c] - mx);
  float ls = logf(s);
#pragma unroll
  for (int c = 0; c < 5; ++c) out[bi * 5 + c] = l[c] - mx - ls;
}

// ---------------------------------------------------------------------------
extern "C" void kernel_launch(void* const* d_in, const int* in_sizes, int n_in,
                              void* d_out, int out_size, void* d_ws, size_t ws_size,
                              hipStream_t stream) {
  const float* x     = (const float*)d_in[0];
  const int*   esrc  = (const int*)d_in[1];
  const int*   edst  = (const int*)d_in[2];
  const int*   etyp  = (const int*)d_in[3];
  const int*   uidx  = (const int*)d_in[4];
  const int*   iidx  = (const int*)d_in[5];
  const float* W1    = (const float*)d_in[22];
  const float* b1    = (const float*)d_in[23];
  const float* W2    = (const float*)d_in[24];
  const float* b2    = (const float*)d_in[25];

  const int N = in_sizes[0] / 4;        // 100000 (divisible by 16)
  const int E = in_sizes[1];            // 1600000
  const int B = in_sizes[4];            // 4096
  float* out = (float*)d_out;

  // Workspace layout (256B-aligned offsets)
  char* ws = (char*)d_ws;
  _Float16* hf16    = (_Float16*)(ws);                    //  6.4 MB
  _Float16* hr      = (_Float16*)(ws + 6400000);          // 32.0 MB
  float*    aggLoop = (float*)   (ws + 38400000);         // 12.8 MB
  _Float16* agg16   = (_Float16*)(ws + 51200000);         //  6.4 MB
  float*    concat  = (float*)   (ws + 57600000);         // 51.2 MB
  _Float16* Wf16    = (_Float16*)(ws + 108800000);        // 12 KB
  _Float16* W1t     = (_Float16*)(ws + 108812288);        // 64 KB
  _Float16* zf16    = (_Float16*)(ws + 108877824);        //  2 MB
  _Float16* z1f16   = (_Float16*)(ws + 110974976);        //  1 MB

  igmc_init_h0<<<(N * 32 + 255) / 256, 256, 0, stream>>>(x, hf16, agg16, N);
  igmc_cvt_w1<<<128, 256, 0, stream>>>(W1, W1t);

  for (int l = 0; l < 4; ++l) {
    const float* bases = (const float*)d_in[6 + 4 * l];
    const float* comp  = (const float*)d_in[7 + 4 * l];
    const float* loopw = (const float*)d_in[8 + 4 * l];
    const float* bias  = (const float*)d_in[9 + 4 * l];
    int Kin = (l == 0) ? 4 : 32;

    igmc_prep_weights<<<24, 256, 0, stream>>>(bases, comp, loopw, Kin, Wf16);

    int waves = N >> 4;                          // 6250 row tiles
    igmc_gemm_hr<<<(waves + 7) / 8, 256, 0, stream>>>(hf16, Wf16, bias, hr,
                                                      aggLoop, N);

    long long et = (long long)E * 4;
    igmc_edge_scatter<<<(int)((et + 255) / 256), 256, 0, stream>>>(
        esrc, edst, etyp, hr, agg16, E, N);

    igmc_activate<<<(N * 32 + 255) / 256, 256, 0, stream>>>(aggLoop, agg16,
                                                            concat, hf16, l, N);
  }

  igmc_gather_z<<<(B * 256 + 255) / 256, 256, 0, stream>>>(concat, uidx, iidx, zf16, B);

  int mlpWaves = (B >> 4) * 8;                    // 2048
  igmc_mlp1<<<(mlpWaves + 7) / 8, 256, 0, stream>>>(zf16, W1t, b1, z1f16, B);

  igmc_head<<<(B + 255) / 256, 256, 0, stream>>>(z1f16, W2, b2, out, B);
}
